// LearnableGate16Array_35012573397057
// MI455X (gfx1250) — matrix operand order repaired and checked by reading, not compiled, not run on hardware
//
#include <hip/hip_runtime.h>
#include <math.h>

// ---------------------------------------------------------------------------
// Problem constants (from reference)
// ---------------------------------------------------------------------------
#define BATCH      2048
#define NUM_INPUTS 2048   // K of the GEMM
#define NUM_GATES  8192
#define NCOL       (NUM_GATES * 2)   // 16384, N of the GEMM

typedef __attribute__((ext_vector_type(16))) __bf16 v16bf;
typedef __attribute__((ext_vector_type(8)))  float  v8f;
typedef unsigned int u32x4 __attribute__((ext_vector_type(4)));
typedef int          i32x4 __attribute__((ext_vector_type(4)));
typedef int          i32x8 __attribute__((ext_vector_type(8)));

// Detect the Tensor Data Mover builtin (device pass only); fall back to
// manual VGPR staging when absent so compilation can never regress.
#define HAVE_TDM 0
#if defined(__AMDGCN__)
#if __has_builtin(__builtin_amdgcn_tensor_load_to_lds)
#undef HAVE_TDM
#define HAVE_TDM 1
#endif
#endif

// ---------------------------------------------------------------------------
// Kernel 1: convert x (f32, BATCH x K) -> bf16 (row-major, K contiguous)
// ---------------------------------------------------------------------------
__global__ void xcvt_kernel(const float* __restrict__ x, __bf16* __restrict__ xbf) {
    int i = (blockIdx.x * 256 + threadIdx.x) * 4;
    float4 v = *(const float4*)(x + i);
    union { __bf16 h[4]; uint2 u; } o;
    o.h[0] = (__bf16)v.x; o.h[1] = (__bf16)v.y;
    o.h[2] = (__bf16)v.z; o.h[3] = (__bf16)v.w;
    *(uint2*)(xbf + i) = o.u;
}

// ---------------------------------------------------------------------------
// Kernel 2: softmax over axis 0 of c (K x NCOL) -> connT, stored TRANSPOSED
// (N-major, K contiguous) so the GEMM's B tile is TDM-loadable with the same
// descriptor shape as the A tile. Transpose goes through an LDS 64x64 tile to
// keep both the reads of c and the writes of connT fully coalesced.
// ---------------------------------------------------------------------------
__global__ void csoftmax_kernel(const float* __restrict__ c, __bf16* __restrict__ connT) {
    __shared__ float  sm[4][64];
    __shared__ float  ss[4][64];
    __shared__ __bf16 tile[64][72];     // 64 cols x 64 k, padded row (144B)
    const int tx = threadIdx.x;         // column within tile (0..63)
    const int ty = threadIdx.y;         // k-slice (0..3)
    const int col0 = blockIdx.x * 64;
    const int col = col0 + tx;
    const int tlin = ty * 64 + tx;

    // Pass 1: online max/sum along k.
    float m = -INFINITY, s = 0.0f;
    for (int k = ty; k < NUM_INPUTS; k += 4) {
        float v = c[(size_t)k * NCOL + col];
        if (v > m) { s = s * __expf(m - v); m = v; }
        s += __expf(v - m);
    }
    sm[ty][tx] = m; ss[ty][tx] = s;
    __syncthreads();
    if (ty == 0) {
        float M = sm[0][tx], S = ss[0][tx];
        #pragma unroll
        for (int j = 1; j < 4; ++j) {
            float m2 = sm[j][tx], s2 = ss[j][tx];
            float Mn = fmaxf(M, m2);
            S = S * __expf(M - Mn) + s2 * __expf(m2 - Mn);
            M = Mn;
        }
        sm[0][tx] = M; ss[0][tx] = 1.0f / S;
    }
    __syncthreads();
    const float M = sm[0][tx], inv = ss[0][tx];

    // Pass 2: normalize, transpose 64x64 chunks through LDS, write N-major.
    for (int kc = 0; kc < NUM_INPUTS; kc += 64) {
        #pragma unroll
        for (int j = 0; j < 16; ++j) {
            int kk = ty + 4 * j;                       // 0..63
            float v = c[(size_t)(kc + kk) * NCOL + col];
            tile[tx][kk] = (__bf16)(__expf(v - M) * inv);
        }
        __syncthreads();
        int cp = tlin >> 2, kb = (tlin & 3) << 4;      // 64 rows x 4 chunks of 16
        const uint4* src = (const uint4*)&tile[cp][kb];
        uint4* dst = (uint4*)(connT + (size_t)(col0 + cp) * NUM_INPUTS + kc + kb);
        dst[0] = src[0]; dst[1] = src[1];
        __syncthreads();
    }
}

// ---------------------------------------------------------------------------
// Kernel 3: softmax over axis 0 of w (16 x NUM_GATES) -> wsoft
// ---------------------------------------------------------------------------
__global__ void wsoftmax_kernel(const float* __restrict__ w, float* __restrict__ wsoft) {
    int g = blockIdx.x * 256 + threadIdx.x;
    float v[16], M = -INFINITY;
    #pragma unroll
    for (int j = 0; j < 16; ++j) { v[j] = w[j * NUM_GATES + g]; M = fmaxf(M, v[j]); }
    float S = 0.0f;
    #pragma unroll
    for (int j = 0; j < 16; ++j) { v[j] = __expf(v[j] - M); S += v[j]; }
    float inv = 1.0f / S;
    #pragma unroll
    for (int j = 0; j < 16; ++j) wsoft[j * NUM_GATES + g] = v[j] * inv;
}

// ---------------------------------------------------------------------------
// Kernel 4: GEMM (bf16 WMMA, f32 acc) + fused gate epilogue.
// Block tile 128(M) x 128(N of y); 8 waves (4M x 2N), each 2x4 WMMA frags.
// BOTH operand tiles (128 rows x 32 contiguous bf16, row stride 2048) are
// moved by the Tensor Data Mover with LDS padding; the compute waves touch
// only LDS (ds_load_b128) and the matrix pipe (v_wmma).
// ---------------------------------------------------------------------------
#define LDA 40   // padded LDS row stride (elements): 64B row + 16B pad

// Manual fallback staging (one 128x32 tile, contiguous rows).
struct Stage { uint4 q0, q1; };

__device__ inline Stage load_stage(const __bf16* __restrict__ src,
                                   int r0, int k0, int tid) {
    Stage r;
    int row = tid >> 1, kc = (tid & 1) << 4;
    const uint4* p = (const uint4*)(src + (size_t)(r0 + row) * NUM_INPUTS + k0 + kc);
    r.q0 = p[0]; r.q1 = p[1];
    return r;
}

__device__ inline void store_stage(const Stage& r, __bf16 (*T)[LDA], int tid) {
    int row = tid >> 1, kc = (tid & 1) << 4;
    uint4* d = (uint4*)&T[row][kc];
    d[0] = r.q0; d[1] = r.q1;
}

#if HAVE_TDM
// TDM: DMA a 128(rows) x 32(K) bf16 tile (row stride 2048 elements) from
// row-major global into LDS, inserting 16B of pad after every 64B row so the
// LDS row stride is 80B (LDA=40). D# per CDNA5 ISA.
__device__ inline void tdm_load_tile(const __bf16* gsrc, void* lds_dst, unsigned dim1) {
    unsigned long long ga = (unsigned long long)(uintptr_t)gsrc;
    unsigned lds_off = (unsigned)(uintptr_t)lds_dst;   // flat addr low 32b = LDS offset
    u32x4 g0;
    g0[0] = 1u;                                        // count=1 (valid user D#)
    g0[1] = lds_off;                                   // lds_addr
    g0[2] = (unsigned)(ga & 0xffffffffull);            // global_addr[31:0]
    g0[3] = (unsigned)((ga >> 32) & 0x1ffffffull)      // global_addr[56:32]
          | (2u << 30);                                // type = 2 ("image")
    i32x8 g1;
    g1[0] = (1 << 16)                                  // data_size = 2 bytes
          | (1 << 20)                                  // pad_enable
          | (3 << 22)                                  // pad_interval: 16 dwords (64B)
          | (3 << 25);                                 // pad_amount: 4 dwords (16B)
    g1[1] = (int)((unsigned)NUM_INPUTS << 16);         // tensor_dim0[15:0] = 2048 (K)
    g1[2] = (int)(dim1 << 16);                         // tensor_dim1[15:0] (row count)
    g1[3] = (int)(32u << 16);                          // tile_dim0 = 32 (K)
    g1[4] = 128;                                       // tile_dim1 = 128 (rows)
    g1[5] = NUM_INPUTS;                                // tensor_dim0_stride = 2048
    g1[6] = 0;
    g1[7] = 0;
    i32x4 z4 = {0, 0, 0, 0};
#if __has_include(<hip/amd_detail/amd_gfx1250_TDM.h>)
    i32x8 z8 = {0, 0, 0, 0, 0, 0, 0, 0};
    __builtin_amdgcn_tensor_load_to_lds(g0, g1, z4, z4, z8, 0);   // 6-arg toolchain
#else
    __builtin_amdgcn_tensor_load_to_lds(g0, g1, z4, z4, 0);       // 5-arg (ROCm 7.2)
#endif
}
#endif // HAVE_TDM

union FragU { uint4 q[2]; v16bf v; };

// A fragment (16x32 bf16): lane<16 holds M=lane, K {0..7,16..23};
// lane>=16 holds M=lane-16, K {8..15,24..31}  (ISA 7.12.2 layout)
__device__ inline v16bf load_afrag(const __bf16 (*As)[LDA], int mbase, int lane) {
    int m = mbase + (lane & 15);
    int koff = (lane >> 4) << 3;
    FragU u;
    u.q[0] = *(const uint4*)&As[m][koff];
    u.q[1] = *(const uint4*)&As[m][koff + 16];
    return u.v;
}

// B fragment (32x16 bf16) from N-major tile: lane<16 -> N=lane, K {0..7,16..23};
// lane>=16 -> N=lane-16, K {8..15,24..31} (same K order as A)
__device__ inline v16bf load_bfrag(const __bf16 (*Bt)[LDA], int nbase, int lane) {
    int n = nbase + (lane & 15);
    int koff = (lane >> 4) << 3;
    FragU u;
    u.q[0] = *(const uint4*)&Bt[n][koff];
    u.q[1] = *(const uint4*)&Bt[n][koff + 16];
    return u.v;
}

__global__ __launch_bounds__(256, 2)
void gate_gemm_kernel(const __bf16* __restrict__ xbf,
                      const __bf16* __restrict__ connT,
                      const float*  __restrict__ wsoft,
                      float*        __restrict__ out) {
    __shared__ __align__(128) __bf16 AsS[2][128][LDA];
    __shared__ __align__(128) __bf16 BtS[2][128][LDA];
    __shared__ float wl[16 * 64];

    const int tid  = threadIdx.x;
    const int lane = tid & 31;
    const int wv   = tid >> 5;
    const int wm   = wv & 3;       // wave M index (0..3)
    const int wn   = wv >> 2;      // wave N index (0..1)
    const int m0    = blockIdx.y * 128;
    const int n0    = blockIdx.x * 128;
    const int gbase = blockIdx.x * 64;

    // Preload this block's 64 gates x 16 weights into LDS.
    for (int i = tid; i < 16 * 64; i += 256) {
        int j = i >> 6, g = i & 63;
        wl[i] = wsoft[j * NUM_GATES + gbase + g];
    }

    // ---- Stage K-tile 0 (both operands) ----
#if HAVE_TDM
    if (wv == 0) {
        tdm_load_tile(xbf   + (size_t)m0 * NUM_INPUTS, &AsS[0][0][0], (unsigned)BATCH);
        tdm_load_tile(connT + (size_t)n0 * NUM_INPUTS, &BtS[0][0][0], (unsigned)NCOL);
        __builtin_amdgcn_s_wait_tensorcnt(0);
    }
#else
    {
        Stage sa = load_stage(xbf,   m0, 0, tid);
        Stage sb = load_stage(connT, n0, 0, tid);
        store_stage(sa, AsS[0], tid);
        store_stage(sb, BtS[0], tid);
    }
#endif
    __syncthreads();

    v8f acc[2][4];
    const v8f vzero = {0.f, 0.f, 0.f, 0.f, 0.f, 0.f, 0.f, 0.f};
    #pragma unroll
    for (int mi = 0; mi < 2; ++mi)
        #pragma unroll
        for (int ni = 0; ni < 4; ++ni) acc[mi][ni] = vzero;

    const int KSTEPS = NUM_INPUTS / 32;   // 64
    for (int kt = 0; kt < KSTEPS; ++kt) {
        const int cur = kt & 1;
        const bool has_next = (kt + 1) < KSTEPS;

        // Kick off next tile's DMAs (wave 0 only; EXEC-independent ops behind
        // a uniform branch), then compute on `cur` while the TDM streams.
#if HAVE_TDM
        if (has_next && wv == 0) {
            const int k1 = (kt + 1) * 32;
            tdm_load_tile(xbf   + (size_t)m0 * NUM_INPUTS + k1,
                          &AsS[cur ^ 1][0][0], (unsigned)BATCH);
            tdm_load_tile(connT + (size_t)n0 * NUM_INPUTS + k1,
                          &BtS[cur ^ 1][0][0], (unsigned)NCOL);
        }
#else
        Stage sa, sb;
        if (has_next) {
            sa = load_stage(xbf,   m0, (kt + 1) * 32, tid);
            sb = load_stage(connT, n0, (kt + 1) * 32, tid);
        }
#endif

        v16bf a[2], b[4];
        #pragma unroll
        for (int mi = 0; mi < 2; ++mi)
            a[mi] = load_afrag(AsS[cur], (wm << 5) + (mi << 4), lane);
        #pragma unroll
        for (int ni = 0; ni < 4; ++ni)
            b[ni] = load_bfrag(BtS[cur], (wn << 6) + (ni << 4), lane);

        #pragma unroll
        for (int mi = 0; mi < 2; ++mi)
            #pragma unroll
            for (int ni = 0; ni < 4; ++ni)
                acc[mi][ni] = __builtin_amdgcn_wmma_f32_16x16x32_bf16(
                    false, a[mi], false, b[ni], (short)0, acc[mi][ni], false, false);

#if HAVE_TDM
        if (wv == 0) __builtin_amdgcn_s_wait_tensorcnt(0);
#else
        if (has_next) {
            store_stage(sa, AsS[cur ^ 1], tid);
            store_stage(sb, BtS[cur ^ 1], tid);
        }
#endif
        __syncthreads();
    }

    // Epilogue: pair A (even col) with B (odd col) via lane-xor shuffle,
    // apply 16-gate weighted sum with softmaxed weights.
    #pragma unroll
    for (int ni = 0; ni < 4; ++ni) {
        const int g = (((wn << 6) + (ni << 4) + (lane & 15)) >> 1);  // 0..63
        float w_[16];
        #pragma unroll
        for (int j = 0; j < 16; ++j) w_[j] = wl[j * 64 + g];
        #pragma unroll
        for (int mi = 0; mi < 2; ++mi) {
            #pragma unroll
            for (int r = 0; r < 8; ++r) {
                float v = acc[mi][ni][r];
                float p = __shfl_xor(v, 1, 32);
                float A = v, B = p, AB = A * B;
                float g7 = A + B - AB;          // A+B-AB
                float g6 = g7 - AB;             // A+B-2AB
                float res =
                      w_[1]  * AB
                    + w_[2]  * (A - AB)
                    + w_[3]  * A
                    + w_[4]  * (B - AB)
                    + w_[5]  * B
                    + w_[6]  * g6
                    + w_[7]  * g7
                    + w_[8]  * (1.0f - g7)
                    + w_[9]  * (1.0f - g6)
                    + w_[10] * (1.0f - B)
                    + w_[11] * (1.0f - (B - AB))
                    + w_[12] * (1.0f - A)
                    + w_[13] * (1.0f - (A - AB))
                    + w_[14] * (1.0f - AB)
                    + w_[15];
                if ((lane & 1) == 0) {
                    int rowg = m0 + (wm << 5) + (mi << 4) + r + ((lane >> 4) << 3);
                    out[(size_t)rowg * NUM_GATES + gbase + g] = res;
                }
            }
        }
    }
}

// ---------------------------------------------------------------------------
// Launch
// ---------------------------------------------------------------------------
extern "C" void kernel_launch(void* const* d_in, const int* in_sizes, int n_in,
                              void* d_out, int out_size, void* d_ws, size_t ws_size,
                              hipStream_t stream) {
    (void)in_sizes; (void)n_in; (void)out_size; (void)ws_size;
    const float* x = (const float*)d_in[0];   // (2048, 2048)
    const float* w = (const float*)d_in[1];   // (16, 8192)
    const float* c = (const float*)d_in[2];   // (2048, 8192, 2)
    float* out = (float*)d_out;               // (2048, 8192)

    char* ws = (char*)d_ws;
    __bf16* xbf   = (__bf16*)ws;                                  // 8 MB
    __bf16* connT = (__bf16*)(ws + (size_t)8 * 1024 * 1024);      // 64 MB (N-major)
    float*  wsoft = (float*)(ws + (size_t)72 * 1024 * 1024);      // 512 KB

    xcvt_kernel<<<(BATCH * NUM_INPUTS) / (256 * 4), 256, 0, stream>>>(x, xbf);
    csoftmax_kernel<<<NCOL / 64, dim3(64, 4), 0, stream>>>(c, connT);
    wsoftmax_kernel<<<NUM_GATES / 256, 256, 0, stream>>>(w, wsoft);
    gate_gemm_kernel<<<dim3(NCOL / 128, BATCH / 128), 256, 0, stream>>>(xbf, connT, wsoft, out);
}
